// HHfan_13795434954859
// MI455X (gfx1250) — matrix operand order repaired and checked by reading, not compiled, not run on hardware
//
#include <hip/hip_runtime.h>

typedef __attribute__((ext_vector_type(16))) _Float16 v16h;
typedef __attribute__((ext_vector_type(8)))  float    v8f;

#define NNODES 30000

__device__ __forceinline__ v16h ldfrag(const _Float16* p0, const _Float16* p1) {
  union { v16h v; uint4 q[2]; } u;
  u.q[0] = *(const uint4*)p0;
  u.q[1] = *(const uint4*)p1;
  return u.v;
}

// native CDNA5 transcendental
__device__ __forceinline__ float tanh_hw(float x) {
  float y;
  asm("v_tanh_f32 %0, %1" : "=v"(y) : "v"(x));
  return y;
}
__device__ __forceinline__ float sig_hw(float x) {
  return 0.5f * tanh_hw(0.5f * x) + 0.5f;
}

__global__ void cvt_f16(const float* __restrict__ s, _Float16* __restrict__ d, int n) {
  int i = blockIdx.x * blockDim.x + threadIdx.x;
  if (i < n) d[i] = (_Float16)s[i];
}

// Fused bidirectional LSTM + mean over sequence, f16 inputs, f16 WMMA, f32 accum.
// Block = 256 threads (8 waves) handling a tile of 16 nodes.
// waves 0-3: forward direction, waves 4-7: backward.
// Wave w (within a direction) owns hidden channels [16w, 16w+16) of each gate.
template<int S>
__global__ __launch_bounds__(256, 2)
void bilstm_mean(const _Float16* __restrict__ x,    // f16 feats [N,S,128] or f16 c-table [N,128]
                 const int*   __restrict__ idx,     // [N,S] gather indices, or nullptr
                 const _Float16* __restrict__ wih,  // [2,256,128] f16, row = gate col
                 const _Float16* __restrict__ whh,  // [2,256,64]  f16
                 const float* __restrict__ bias,    // [2,256]
                 float* __restrict__ out,           // [N,128] (0:64 fwd mean, 64:128 bwd)
                 _Float16* __restrict__ out16)      // optional f16 copy of out
{
  __shared__ _Float16 xl[16 * S][136];   // f16 x tile, padded row stride (272 B)
  __shared__ _Float16 hl[2][16][72];     // per-direction h state [node][channel]

  const int tid   = threadIdx.x;
  const int node0 = blockIdx.x * 16;

  // ---- stage x tile into LDS via async global->LDS copies (16 B chunks) ----
  for (int i = tid; i < 16 * S * 16; i += 256) {
    int row = i >> 4;                 // node*S + s
    int ch  = i & 15;                 // 16-byte chunk within the 256 B row
    int nd  = row / S, s = row - nd * S;
    const _Float16* src = idx
      ? (x + (size_t)idx[(node0 + nd) * S + s] * 128)
      : (x + ((size_t)(node0 + nd) * S + s) * 128);
    unsigned long long gaddr = (unsigned long long)(uintptr_t)(src + ch * 8);
    unsigned int       laddr = (unsigned int)(uintptr_t)&xl[row][ch * 8];
    asm volatile("global_load_async_to_lds_b128 %0, %1, off"
                 :: "v"(laddr), "v"(gaddr) : "memory");
  }
  for (int i = tid; i < 2 * 16 * 72; i += 256) ((_Float16*)hl)[i] = (_Float16)0.f;
  asm volatile("s_wait_asynccnt 0" ::: "memory");
  __syncthreads();

  const int lane = tid & 31;
  const int wave = tid >> 5;
  const int dir  = wave >> 2;          // 0 = fwd, 1 = bwd
  const int wsub = wave & 3;           // 16-channel block within each gate
  const int half = (lane < 16) ? 0 : 1;
  const int mrow = lane & 15;

  // Hoist Whh B-fragments (loop invariant): [gate][K-chunk of 32]
  v16h whB[4][2];
  #pragma unroll
  for (int g = 0; g < 4; ++g)
    #pragma unroll
    for (int kc = 0; kc < 2; ++kc) {
      const _Float16* p = whh + ((size_t)(dir * 256 + g * 64 + wsub * 16 + mrow) * 64
                                 + kc * 32 + half * 8);
      whB[g][kc] = ldfrag(p, p + 16);
    }

  float bg[4];
  #pragma unroll
  for (int g = 0; g < 4; ++g)
    bg[g] = bias[dir * 256 + g * 64 + wsub * 16 + mrow];

  v8f cstate = {};
  v8f hsum   = {};

  for (int t = 0; t < S; ++t) {
    const int s = dir ? (S - 1 - t) : t;

    // A-fragments from x tile (K=128 -> 4 chunks of 32)
    v16h ax[4];
    #pragma unroll
    for (int kc = 0; kc < 4; ++kc) {
      const _Float16* p = &xl[mrow * S + s][kc * 32 + half * 8];
      ax[kc] = ldfrag(p, p + 16);
    }
    // A-fragments from h state (K=64 -> 2 chunks)
    v16h ah[2];
    #pragma unroll
    for (int kc = 0; kc < 2; ++kc) {
      const _Float16* p = &hl[dir][mrow][kc * 32 + half * 8];
      ah[kc] = ldfrag(p, p + 16);
    }

    v8f acc[4];
    #pragma unroll
    for (int g = 0; g < 4; ++g) {
      v8f a = { bg[g], bg[g], bg[g], bg[g], bg[g], bg[g], bg[g], bg[g] };
      #pragma unroll
      for (int kc = 0; kc < 4; ++kc) {
        const _Float16* p = wih + ((size_t)(dir * 256 + g * 64 + wsub * 16 + mrow) * 128
                                   + kc * 32 + half * 8);
        v16h bfrag = ldfrag(p, p + 16);
        a = __builtin_amdgcn_wmma_f32_16x16x32_f16(false, ax[kc], false, bfrag,
                                                   (short)0, a, false, false);
      }
      #pragma unroll
      for (int kc = 0; kc < 2; ++kc)
        a = __builtin_amdgcn_wmma_f32_16x16x32_f16(false, ah[kc], false, whB[g][kc],
                                                   (short)0, a, false, false);
      acc[g] = a;
    }

    // LSTM cell nonlinearity (native v_tanh_f32 for all four gates)
    float hv[8];
    #pragma unroll
    for (int j = 0; j < 8; ++j) {
      float iv = sig_hw(acc[0][j]);
      float fv = sig_hw(acc[1][j]);
      float gv = tanh_hw(acc[2][j]);
      float ov = sig_hw(acc[3][j]);
      float cn = fv * cstate[j] + iv * gv;
      cstate[j] = cn;
      hv[j] = ov * tanh_hw(cn);
      hsum[j] += hv[j];
    }

    __syncthreads();   // all waves done reading previous h
    #pragma unroll
    for (int j = 0; j < 8; ++j)
      hl[dir][j + half * 8][wsub * 16 + mrow] = (_Float16)hv[j];
    __syncthreads();   // new h visible
  }

  const float inv = 1.0f / (float)S;
  #pragma unroll
  for (int j = 0; j < 8; ++j) {
    float m = hsum[j] * inv;
    size_t o = (size_t)(node0 + j + half * 8) * 128 + dir * 64 + wsub * 16 + mrow;
    out[o] = m;
    if (out16) out16[o] = (_Float16)m;
  }
}

// Attention combine: one wave per node.
__global__ void combine_kernel(const float* __restrict__ c,
                               const float* __restrict__ n0,
                               const float* __restrict__ n1,
                               const float* __restrict__ attn,   // [256]
                               float* __restrict__ out, int nnodes)
{
  int wave = threadIdx.x >> 5, lane = threadIdx.x & 31;
  int node = blockIdx.x * 8 + wave;
  if (node >= nnodes) return;
  const float* cp = c  + (size_t)node * 128;
  const float* p0 = n0 + (size_t)node * 128;
  const float* p1 = n1 + (size_t)node * 128;
  float a1[4], a2[4], cv[4], v0[4], v1[4];
  float r0 = 0.f, r1 = 0.f, r2 = 0.f, r3 = 0.f;   // a1·c, a2·c, a2·n0, a2·n1
  #pragma unroll
  for (int j = 0; j < 4; ++j) {
    int e = lane + 32 * j;
    a1[j] = attn[e]; a2[j] = attn[128 + e];
    cv[j] = cp[e]; v0[j] = p0[e]; v1[j] = p1[e];
    r0 += a1[j] * cv[j];
    r1 += a2[j] * cv[j];
    r2 += a2[j] * v0[j];
    r3 += a2[j] * v1[j];
  }
  #pragma unroll
  for (int m = 16; m >= 1; m >>= 1) {
    r0 += __shfl_xor(r0, m, 32);
    r1 += __shfl_xor(r1, m, 32);
    r2 += __shfl_xor(r2, m, 32);
    r3 += __shfl_xor(r3, m, 32);
  }
  float ss = r0 + r1, s0 = r0 + r2, s1 = r0 + r3;
  ss = ss > 0.f ? ss : 0.01f * ss;
  s0 = s0 > 0.f ? s0 : 0.01f * s0;
  s1 = s1 > 0.f ? s1 : 0.01f * s1;
  float mx = fmaxf(ss, fmaxf(s0, s1));
  float w0 = __expf(ss - mx), w1 = __expf(s0 - mx), w2 = __expf(s1 - mx);
  float inv = 1.f / (w0 + w1 + w2);
  w0 *= inv; w1 *= inv; w2 *= inv;
  float* op = out + (size_t)node * 128;
  #pragma unroll
  for (int j = 0; j < 4; ++j)
    op[lane + 32 * j] = w0 * cv[j] + w1 * v0[j] + w2 * v1[j];
}

extern "C" void kernel_launch(void* const* d_in, const int* in_sizes, int n_in,
                              void* d_out, int out_size, void* d_ws, size_t ws_size,
                              hipStream_t stream) {
  const float* feats_a = (const float*)d_in[0];
  const float* feats_b = (const float*)d_in[1];
  const float* Wih_ca = (const float*)d_in[2];
  const float* Whh_ca = (const float*)d_in[3];
  const float* b_ca   = (const float*)d_in[4];
  const float* Wih_cb = (const float*)d_in[5];
  const float* Whh_cb = (const float*)d_in[6];
  const float* b_cb   = (const float*)d_in[7];
  const float* Wih_na = (const float*)d_in[8];
  const float* Whh_na = (const float*)d_in[9];
  const float* b_na   = (const float*)d_in[10];
  const float* Wih_nb = (const float*)d_in[11];
  const float* Whh_nb = (const float*)d_in[12];
  const float* b_nb   = (const float*)d_in[13];
  const float* attn_a = (const float*)d_in[14];
  const float* attn_b = (const float*)d_in[15];
  const int* idx_a_a = (const int*)d_in[16];
  const int* idx_b_a = (const int*)d_in[17];
  const int* idx_a_b = (const int*)d_in[18];
  const int* idx_b_b = (const int*)d_in[19];

  // workspace layout
  char* ws = (char*)d_ws;
  const int WIH_N = 2 * 256 * 128, WHH_N = 2 * 256 * 64;
  const int FEAT_N = NNODES * 5 * 128;          // 19.2M elements
  const int CTAB_N = NNODES * 128;              // 3.84M elements
  _Float16* wih16[4]; _Float16* whh16[4];
  size_t off = 0;
  for (int i = 0; i < 4; ++i) { wih16[i] = (_Float16*)(ws + off); off += (size_t)WIH_N * 2; }
  for (int i = 0; i < 4; ++i) { whh16[i] = (_Float16*)(ws + off); off += (size_t)WHH_N * 2; }
  _Float16* f16_a = (_Float16*)(ws + off); off += (size_t)FEAT_N * 2;
  _Float16* f16_b = (_Float16*)(ws + off); off += (size_t)FEAT_N * 2;
  _Float16* c16_a = (_Float16*)(ws + off); off += (size_t)CTAB_N * 2;
  _Float16* c16_b = (_Float16*)(ws + off); off += (size_t)CTAB_N * 2;
  const size_t VEC = (size_t)NNODES * 128 * 4;
  float* c_a  = (float*)(ws + off); off += VEC;
  float* c_b  = (float*)(ws + off); off += VEC;
  float* n_aa = (float*)(ws + off); off += VEC;
  float* n_ba = (float*)(ws + off); off += VEC;
  float* n_ab = (float*)(ws + off); off += VEC;
  float* n_bb = (float*)(ws + off); off += VEC;

  // f32 -> f16 conversions (weights + feature streams)
  const float* wihsrc[4] = {Wih_ca, Wih_cb, Wih_na, Wih_nb};
  const float* whhsrc[4] = {Whh_ca, Whh_cb, Whh_na, Whh_nb};
  for (int i = 0; i < 4; ++i) {
    cvt_f16<<<dim3((WIH_N + 255) / 256), dim3(256), 0, stream>>>(wihsrc[i], wih16[i], WIH_N);
    cvt_f16<<<dim3((WHH_N + 255) / 256), dim3(256), 0, stream>>>(whhsrc[i], whh16[i], WHH_N);
  }
  cvt_f16<<<dim3((FEAT_N + 255) / 256), dim3(256), 0, stream>>>(feats_a, f16_a, FEAT_N);
  cvt_f16<<<dim3((FEAT_N + 255) / 256), dim3(256), 0, stream>>>(feats_b, f16_b, FEAT_N);

  dim3 gB(NNODES / 16), tB(256);
  // content aggregation (S = L = 5); writes f32 c and f16 c-table for the gathers
  bilstm_mean<5><<<gB, tB, 0, stream>>>(f16_a, nullptr, wih16[0], whh16[0], b_ca, c_a, c16_a);
  bilstm_mean<5><<<gB, tB, 0, stream>>>(f16_b, nullptr, wih16[1], whh16[1], b_cb, c_b, c16_b);
  // neighbor aggregation (S = K = 16, gather from f16 c tables; L2-resident)
  bilstm_mean<16><<<gB, tB, 0, stream>>>(c16_a, idx_a_a, wih16[2], whh16[2], b_na, n_aa, nullptr);
  bilstm_mean<16><<<gB, tB, 0, stream>>>(c16_b, idx_b_a, wih16[3], whh16[3], b_nb, n_ba, nullptr);
  bilstm_mean<16><<<gB, tB, 0, stream>>>(c16_a, idx_a_b, wih16[2], whh16[2], b_na, n_ab, nullptr);
  bilstm_mean<16><<<gB, tB, 0, stream>>>(c16_b, idx_b_b, wih16[3], whh16[3], b_nb, n_bb, nullptr);

  float* out_a = (float*)d_out;
  float* out_b = out_a + (size_t)NNODES * 128;
  combine_kernel<<<dim3((NNODES + 7) / 8), dim3(256), 0, stream>>>(c_a, n_aa, n_ba, attn_a, out_a, NNODES);
  combine_kernel<<<dim3((NNODES + 7) / 8), dim3(256), 0, stream>>>(c_b, n_ab, n_bb, attn_b, out_b, NNODES);
}
